// EmbeddingSoftAttention_21603685499065
// MI455X (gfx1250) — matrix operand order repaired and checked by reading, not compiled, not run on hardware
//
#include <hip/hip_runtime.h>
#include <hip/hip_bf16.h>

// ---------------------------------------------------------------------------
// EmbeddingSoftAttention for MI455X (gfx1250, wave32, WMMA 16x16x32 f16)
//
// out = sigmoid( mean_l( relu(embs[context] @ W1 + b1) ) @ W2 + b2 ) * embs[x]
//
// GEMM1 is 31.5 GFLOP; gather traffic 245MB (table=120MB < 192MB L2).
// f16 WMMA for both GEMMs. k1: LDS-staged gathered A-tiles (double-buffered,
// one barrier/stage), B-fragments and A-fragments fully register-resident so
// the 10 WMMAs issue back-to-back behind a single s_wait_dscnt.
// ---------------------------------------------------------------------------

typedef __attribute__((ext_vector_type(8)))  _Float16 v8h;
typedef __attribute__((ext_vector_type(16))) _Float16 v16h;
typedef __attribute__((ext_vector_type(8)))  float    v8f;

#define NTOKENS 100000
#define NCOND   300
#define KPAD1   320     // NCOND padded to multiple of 32
#define NHID    256
#define BATCH   4096
#define CTXLEN  50
#define NPAD2   304     // NCOND padded to multiple of 16 (N dim of GEMM2)

__device__ __forceinline__ v16h h16cat(v8h lo, v8h hi) {
  return __builtin_shufflevector(lo, hi, 0,1,2,3,4,5,6,7,8,9,10,11,12,13,14,15);
}

// ---------------------------------------------------------------------------
// prep: W1 [300][256] f32 -> W1T [256][320] f16 (n-major, K contiguous, 0-pad)
//       W2 [256][300] f32 -> W2T [304][256] f16 (n-major, K contiguous, 0-pad)
// ---------------------------------------------------------------------------
__global__ __launch_bounds__(256)
void prep_weights(const float* __restrict__ W1, const float* __restrict__ W2,
                  _Float16* __restrict__ W1T, _Float16* __restrict__ W2T) {
  int i = blockIdx.x * 256 + threadIdx.x;
  const int n1 = NHID * KPAD1;          // 81920
  const int n2 = NPAD2 * NHID;          // 77824
  if (i < n1) {
    int n = i / KPAD1, k = i % KPAD1;
    W1T[i] = (k < NCOND) ? (_Float16)W1[k * NHID + n] : (_Float16)0.0f;
  } else if (i < n1 + n2) {
    int j = i - n1;
    int n = j / NHID, k = j % NHID;
    W2T[j] = (n < NCOND) ? (_Float16)W2[k * NCOND + n] : (_Float16)0.0f;
  }
}

// ---- k1 staging helpers: one chunk = 8 consecutive f16 of one A-tile row ----
// Tile = 16 rows x 320 f16 -> 640 chunks. Chunk c: row r=c/40, k=(c%40)*8.
// Embedding rows are 16B-aligned (300*4 = 75*16), so float4 loads are legal.
__device__ __forceinline__ void chunk_load(int b, int st, int c,
                                           const int* __restrict__ context,
                                           const float* __restrict__ embs,
                                           float4& lo, float4& hi) {
  lo = make_float4(0.f, 0.f, 0.f, 0.f);
  hi = make_float4(0.f, 0.f, 0.f, 0.f);
  int r  = c / (KPAD1 / 8);
  int kc = (c % (KPAD1 / 8)) * 8;
  int l  = (st & 3) * 16 + r;
  if (l < CTXLEN) {
    int t = context[b * CTXLEN + l];
    const float* ep = embs + (long)t * NCOND;
    if (kc + 8 <= NCOND) {
      lo = *(const float4*)(ep + kc);
      hi = *(const float4*)(ep + kc + 4);
    } else if (kc < NCOND) {            // kc == 296: 4 valid floats
      lo = *(const float4*)(ep + kc);
    }
  }
}

__device__ __forceinline__ void chunk_store(_Float16* __restrict__ buf, int c,
                                            float4 lo, float4 hi) {
  int r  = c / (KPAD1 / 8);
  int kc = (c % (KPAD1 / 8)) * 8;
  v8h h;
  h[0] = (_Float16)lo.x; h[1] = (_Float16)lo.y;
  h[2] = (_Float16)lo.z; h[3] = (_Float16)lo.w;
  h[4] = (_Float16)hi.x; h[5] = (_Float16)hi.y;
  h[6] = (_Float16)hi.z; h[7] = (_Float16)hi.w;
  *(v8h*)(buf + r * KPAD1 + kc) = h;
}

// ---------------------------------------------------------------------------
// k1: workgroup = 16 waves / 512 threads = 16 batch rows.
//   64 stages (16 batches x 4 l-tiles). Per stage: double-buffered LDS A-tile;
//   global gather loads issued, WMMA chain on previous tile runs underneath,
//   then f32->f16 convert + ds_store_b128. Wave w = N-tile w of W1T with all
//   10 B-fragments register-resident; acc init = b1 (fused bias); relu +
//   masked column-sum -> mean pool in LDS.
// ---------------------------------------------------------------------------
__global__ __launch_bounds__(512)
void k1_pooled(const int* __restrict__ context, const float* __restrict__ embs,
               const float* __restrict__ b1, const _Float16* __restrict__ W1T,
               _Float16* __restrict__ pooledOut) {
  __shared__ _Float16 aT[2][16 * KPAD1];  // 2 x 10 KB
  __shared__ float    pool[16 * NHID];    // 16 KB

  const int tid  = threadIdx.x;
  const int lane = tid & 31;
  const int wave = tid >> 5;              // = N-tile 0..15
  const int bTile = blockIdx.x;           // 0..255

  for (int i = tid; i < 16 * NHID; i += 512) pool[i] = 0.0f;

  // B layout (32x16): lanes0-15 K=0..7,16..23 ; lanes16-31 K=8..15,24..31
  // -> per k-step two 16B chunks at +0/+16 (lanes<16) or +8/+24.
  const int col = wave * 16 + (lane & 15);            // global N, < 256
  const int kbB = (lane < 16) ? 0 : 8;
  const _Float16* bp = W1T + col * KPAD1 + kbB;
  v16h bfrag[10];
#pragma unroll
  for (int s = 0; s < 10; ++s)
    bfrag[s] = h16cat(*(const v8h*)(bp + s * 32), *(const v8h*)(bp + s * 32 + 16));

  const float b1v   = b1[col];
  const int   kbA   = (lane < 16) ? 0 : 8;            // A layout half-split
  const int   mBase = (lane < 16) ? 0 : 8;            // C layout M offset
  const bool  extra = (tid < 128);                    // chunks 512..639

  // ---- prologue: stage tile 0 into buffer 0 ----
  {
    float4 lo, hi;
    chunk_load(bTile * 16 + 0, 0, tid, context, embs, lo, hi);
    chunk_store(aT[0], tid, lo, hi);
    if (extra) {
      chunk_load(bTile * 16 + 0, 0, tid + 512, context, embs, lo, hi);
      chunk_store(aT[0], tid + 512, lo, hi);
    }
  }
  __syncthreads();

  for (int st = 0; st < 64; ++st) {
    const int bl = st >> 2;               // batch-local row 0..15
    const int lt = st & 3;                // l-tile 0..3
    const _Float16* cur = aT[st & 1];
    _Float16*       nxt = aT[(st + 1) & 1];

    // ---- issue next tile's gather loads (held in regs through compute) ----
    float4 l0, h0, l1, h1;
    const bool haveNext = (st + 1 < 64);
    if (haveNext) {
      int nb = bTile * 16 + ((st + 1) >> 2);
      chunk_load(nb, st + 1, tid, context, embs, l0, h0);
      if (extra) chunk_load(nb, st + 1, tid + 512, context, embs, l1, h1);
    }

    // ---- 16x16x320 WMMA on current tile; all A-fragments preloaded ----
    const _Float16* ap = cur + (lane & 15) * KPAD1 + kbA;
    v16h af[10];
#pragma unroll
    for (int s = 0; s < 10; ++s)
      af[s] = h16cat(*(const v8h*)(ap + s * 32), *(const v8h*)(ap + s * 32 + 16));

    v8f acc;
#pragma unroll
    for (int v = 0; v < 8; ++v) acc[v] = b1v;
#pragma unroll
    for (int s = 0; s < 10; ++s)
      acc = __builtin_amdgcn_wmma_f32_16x16x32_f16(
          false, af[s], false, bfrag[s], (short)0, acc, false, false);

    // ---- convert + store next tile into the other LDS buffer ----
    if (haveNext) {
      chunk_store(nxt, tid, l0, h0);
      if (extra) chunk_store(nxt, tid + 512, l1, h1);
    }

    // ---- relu + masked column-sum (C: lane holds N=lane%16, M=mBase+v) ----
    int nvalid = CTXLEN - lt * 16;
    if (nvalid > 16) nvalid = 16;
    float part = 0.0f;
#pragma unroll
    for (int v = 0; v < 8; ++v) {
      float h = acc[v] > 0.0f ? acc[v] : 0.0f;
      part += ((mBase + v) < nvalid) ? h : 0.0f;
    }
    part += __shfl_xor(part, 16, 32);               // fold M=8..15 half
    if (lane < 16) pool[bl * NHID + col] += part;   // cols disjoint per wave
    __syncthreads();
  }

  for (int i = tid; i < 16 * NHID; i += 512) {
    int r = i >> 8, c = i & 255;
    pooledOut[(long)(bTile * 16 + r) * NHID + c] =
        (_Float16)(pool[i] * (1.0f / (float)CTXLEN));
  }
}

// ---------------------------------------------------------------------------
// k2: one wave per 16x16 output tile.  M=4096 (256 tiles), K=256 (8 steps),
//     N=304 (19 tiles).  All 16 fragments preloaded -> one wait, 8 WMMAs;
//     acc init = b2; fused sigmoid * embs[x[b]][n], f32 store.
// ---------------------------------------------------------------------------
__global__ __launch_bounds__(256)
void k2_gate(const int* __restrict__ x, const float* __restrict__ embs,
             const float* __restrict__ b2, const _Float16* __restrict__ pooled,
             const _Float16* __restrict__ W2T, float* __restrict__ out) {
  const int tid  = threadIdx.x;
  const int lane = tid & 31;
  const int wv   = tid >> 5;
  const int tile = blockIdx.x * 8 + wv;              // uniform per wave
  const int numTiles = (BATCH / 16) * (NPAD2 / 16);  // 4864
  if (tile >= numTiles) return;

  const int mt = tile / (NPAD2 / 16);
  const int nt = tile % (NPAD2 / 16);

  const int kbA  = (lane < 16) ? 0 : 8;
  const int kbB  = (lane < 16) ? 0 : 8;
  const int colL = nt * 16 + (lane & 15);            // 0..303
  const int row  = mt * 16 + (lane & 15);

  const _Float16* ap = pooled + (long)row  * NHID + kbA;
  const _Float16* bp = W2T    + (long)colL * NHID + kbB;

  v16h af[8], bf[8];
#pragma unroll
  for (int s = 0; s < 8; ++s) {
    af[s] = h16cat(*(const v8h*)(ap + s * 32), *(const v8h*)(ap + s * 32 + 16));
    bf[s] = h16cat(*(const v8h*)(bp + s * 32), *(const v8h*)(bp + s * 32 + 16));
  }

  v8f acc;
  const float b2v = (colL < NCOND) ? b2[colL] : 0.0f;
#pragma unroll
  for (int v = 0; v < 8; ++v) acc[v] = b2v;
#pragma unroll
  for (int s = 0; s < 8; ++s)
    acc = __builtin_amdgcn_wmma_f32_16x16x32_f16(
        false, af[s], false, bf[s], (short)0, acc, false, false);

  const int mBase = (lane < 16) ? 0 : 8;
  if (colL < NCOND) {
#pragma unroll
    for (int v = 0; v < 8; ++v) {
      int b = mt * 16 + mBase + v;                   // < 4096
      float g  = 1.0f / (1.0f + __expf(-acc[v]));
      float xe = embs[(long)x[b] * NCOND + colL];
      out[(long)b * NCOND + colL] = g * xe;
    }
  }
}

// ---------------------------------------------------------------------------
extern "C" void kernel_launch(void* const* d_in, const int* in_sizes, int n_in,
                              void* d_out, int out_size, void* d_ws, size_t ws_size,
                              hipStream_t stream) {
  const int*   x       = (const int*)d_in[0];
  const int*   context = (const int*)d_in[1];
  const float* embs    = (const float*)d_in[2];
  const float* W1      = (const float*)d_in[3];
  const float* b1      = (const float*)d_in[4];
  const float* W2      = (const float*)d_in[5];
  const float* b2      = (const float*)d_in[6];
  float*       out     = (float*)d_out;

  // Workspace layout (f16), ~2.4 MB total:
  char* ws = (char*)d_ws;
  _Float16* W1T    = (_Float16*)(ws);                               // 256*320*2 = 163840
  _Float16* W2T    = (_Float16*)(ws + 163840);                      // 304*256*2 = 155648
  _Float16* pooled = (_Float16*)(ws + 163840 + 155648);             // 4096*256*2

  {
    const int total = NHID * KPAD1 + NPAD2 * NHID;                  // 159744
    prep_weights<<<(total + 255) / 256, 256, 0, stream>>>(W1, W2, W1T, W2T);
  }
  k1_pooled<<<BATCH / 16, 512, 0, stream>>>(context, embs, b1, W1T, pooled);
  {
    const int numTiles = (BATCH / 16) * (NPAD2 / 16);               // 4864
    k2_gate<<<numTiles / 8, 256, 0, stream>>>(x, embs, b2, pooled, W2T, out);
  }
}
